// ResidualVectorQuantizer_58480274703092
// MI455X (gfx1250) — matrix operand order repaired and checked by reading, not compiled, not run on hardware
//
#include <hip/hip_runtime.h>

typedef __attribute__((ext_vector_type(16))) __bf16 v16bf;
typedef __attribute__((ext_vector_type(8)))  float  v8f;

#define Bsz   16
#define Dd    256
#define Tt    4096
#define BINS  1024
#define NQ    8
#define NROWS (Bsz * Tt)            // 65536
#define ROWS_PER_BLOCK 128
#define NTILES (BINS / 16)          // 64

// ---------------------------------------------------------------- helpers
__device__ __forceinline__ float waveReduceSum(float s) {
  #pragma unroll
  for (int off = 16; off > 0; off >>= 1) s += __shfl_xor(s, off, 32);
  return s;
}

// Async copy 16B global -> LDS, tracked by ASYNCcnt (CDNA5 ISA 15.18.3 op 98).
__device__ __forceinline__ void async_b128(unsigned int lds_addr, unsigned long long gaddr) {
  asm volatile("global_load_async_to_lds_b128 %0, %1, off"
               :: "v"(lds_addr), "v"(gaddr)
               : "memory");
}
__device__ __forceinline__ void wait_async0() {
  asm volatile("s_wait_asynccnt 0x0" ::: "memory");
}

// A-fragment for V_WMMA_*_16x16x32_bf16 (16x32 MxK):
// lane half h holds K = [kt*32 + h*16, +16) contiguous of its row.
__device__ __forceinline__ v16bf load_a_frag(const float* __restrict__ rp, int kt, int hl) {
  const float4* p0 = (const float4*)(rp + kt * 32 + hl * 16);
  const float4* p1 = (const float4*)(rp + kt * 32 + hl * 16 + 8);
  float4 f0 = p0[0], f1 = p0[1], f2 = p1[0], f3 = p1[1];
  v16bf a;
  a[0]=(__bf16)f0.x;  a[1]=(__bf16)f0.y;  a[2]=(__bf16)f0.z;  a[3]=(__bf16)f0.w;
  a[4]=(__bf16)f1.x;  a[5]=(__bf16)f1.y;  a[6]=(__bf16)f1.z;  a[7]=(__bf16)f1.w;
  a[8]=(__bf16)f2.x;  a[9]=(__bf16)f2.y;  a[10]=(__bf16)f2.z; a[11]=(__bf16)f2.w;
  a[12]=(__bf16)f3.x; a[13]=(__bf16)f3.y; a[14]=(__bf16)f3.z; a[15]=(__bf16)f3.w;
  return a;
}

// ---------------------------------------------------------------- prep
__global__ __launch_bounds__(256) void zero_ws_kernel(float* __restrict__ Qacc,
                                                      float* __restrict__ lossSum) {
  size_t i = (size_t)blockIdx.x * blockDim.x + threadIdx.x;   // 4M float4s
  float4 z = {0.f, 0.f, 0.f, 0.f};
  ((float4*)Qacc)[i] = z;
  if (blockIdx.x == 0 && threadIdx.x < NQ) lossSum[threadIdx.x] = 0.f;
}

// x [B,D,T] -> R [B*T, D]
__global__ __launch_bounds__(256) void transpose_in_kernel(const float* __restrict__ x,
                                                           float* __restrict__ R) {
  __shared__ float tile[32][33];
  const int b = blockIdx.z;
  const int d0 = blockIdx.y * 32, t0 = blockIdx.x * 32;
  const int tx = threadIdx.x, ty = threadIdx.y;
  #pragma unroll
  for (int i = 0; i < 4; ++i) {
    int dy = ty + i * 8;
    tile[dy][tx] = x[((size_t)(b * Dd + d0 + dy)) * Tt + (t0 + tx)];
  }
  __syncthreads();
  #pragma unroll
  for (int i = 0; i < 4; ++i) {
    int dy = ty + i * 8;   // t offset
    R[((size_t)b * Tt + (t0 + dy)) * Dd + (d0 + tx)] = tile[tx][dy];
  }
}

// Qacc [B*T, D] -> out [B,D,T]
__global__ __launch_bounds__(256) void transpose_out_kernel(const float* __restrict__ Qacc,
                                                            float* __restrict__ out) {
  __shared__ float tile[32][33];
  const int b = blockIdx.z;
  const int d0 = blockIdx.y * 32, t0 = blockIdx.x * 32;
  const int tx = threadIdx.x, ty = threadIdx.y;
  #pragma unroll
  for (int i = 0; i < 4; ++i) {
    int dy = ty + i * 8;   // t offset; tx = d offset
    tile[dy][tx] = Qacc[((size_t)b * Tt + (t0 + dy)) * Dd + (d0 + tx)];
  }
  __syncthreads();
  #pragma unroll
  for (int i = 0; i < 4; ++i) {
    int dy = ty + i * 8;   // d offset; tx = t offset
    out[((size_t)(b * Dd + d0 + dy)) * Tt + (t0 + tx)] = tile[tx][dy];
  }
}

// codebooks [NQ*BINS, D] fp32 -> bf16 copy + per-row squared norm
__global__ __launch_bounds__(256) void cb_prep_kernel(const float* __restrict__ cbs,
                                                      unsigned short* __restrict__ cbB_,
                                                      float* __restrict__ enorm) {
  __bf16* cbB = (__bf16*)cbB_;
  const int row = blockIdx.x;
  const int d = threadIdx.x;
  const size_t off = (size_t)row * Dd + d;
  float v = cbs[off];
  cbB[off] = (__bf16)v;
  float s = waveReduceSum(v * v);
  __shared__ float red[8];
  if ((threadIdx.x & 31) == 0) red[threadIdx.x >> 5] = s;
  __syncthreads();
  if (threadIdx.x == 0) {
    float t = 0.f;
    #pragma unroll
    for (int i = 0; i < 8; ++i) t += red[i];
    enorm[row] = t;
  }
}

// ---------------------------------------------------------------- stage: argmin via WMMA
// Block = 256 threads (8 waves) x 128 rows. Codebook tile (16 codes x 256 dims, bf16,
// 8KB) is double-buffered in LDS and filled with async global->LDS copies that overlap
// the WMMA chain of the current tile.
__global__ __launch_bounds__(256) void rvq_argmin_kernel(const float* __restrict__ R,
                                                         const unsigned short* __restrict__ cbB_,
                                                         const float* __restrict__ enorm,
                                                         int* __restrict__ codes) {
  __shared__ __align__(32) __bf16 ldsB[2][16 * Dd];   // 2 x 8KB double buffer
  __shared__ float ensh[BINS];                        // 4KB

  const int tid  = threadIdx.x;
  const int lane = tid & 31;
  const int wave = tid >> 5;
  const int hl = lane >> 4;          // half-wave
  const int m  = lane & 15;
  const int rowBase = blockIdx.x * ROWS_PER_BLOCK + wave * 16;

  const unsigned long long cbBytes = (unsigned long long)(uintptr_t)cbB_;
  unsigned int ldsAddr[2];
  ldsAddr[0] = (unsigned int)(uintptr_t)(&ldsB[0][0]);
  ldsAddr[1] = (unsigned int)(uintptr_t)(&ldsB[1][0]);

  // A fragments: this lane's row, all K=256 (8 tiles of 32), kept resident in VGPRs.
  const float* rp = R + (size_t)(rowBase + m) * Dd;
  v16bf a[8];
  #pragma unroll
  for (int kt = 0; kt < 8; ++kt) a[kt] = load_a_frag(rp, kt, hl);

  // stage ||e||^2 table + first codebook tile
  for (int i = tid; i < BINS; i += 256) ensh[i] = enorm[i];
  {
    unsigned int la = ldsAddr[0] + tid * 32;          // 256 threads x 32B = 8KB
    unsigned long long ga = cbBytes + (unsigned long long)tid * 32;
    async_b128(la, ga);
    async_b128(la + 16, ga + 16);
  }
  wait_async0();
  __syncthreads();

  float minv[8];
  int   mini[8];
  #pragma unroll
  for (int r = 0; r < 8; ++r) { minv[r] = 3.4e38f; mini[r] = 0; }

  for (int nt = 0; nt < NTILES; ++nt) {
    const int buf = nt & 1;
    // prefetch next tile into the other buffer (overlaps this tile's WMMAs)
    if (nt + 1 < NTILES) {
      unsigned int la = ldsAddr[buf ^ 1] + tid * 32;
      unsigned long long ga = cbBytes +
          ((unsigned long long)(nt + 1) * 16 * Dd) * 2ull + (unsigned long long)tid * 32;
      async_b128(la, ga);
      async_b128(la + 16, ga + 16);
    }

    const int n = nt * 16 + m;                        // this lane's codebook row (B column)
    v8f c = {0.f, 0.f, 0.f, 0.f, 0.f, 0.f, 0.f, 0.f};
    #pragma unroll
    for (int kt = 0; kt < 8; ++kt) {
      // B fragment (32x16 KxN): lane half h holds K = [kt*32 + h*16, +16) contiguous bf16
      v16bf bfrag = *(const v16bf*)(&ldsB[buf][(unsigned)m * Dd + hl * 16 + kt * 32]);
      c = __builtin_amdgcn_wmma_f32_16x16x32_bf16(false, a[kt], false, bfrag,
                                                  (short)0, c, false, false);
    }
    const float en = ensh[n];
    #pragma unroll
    for (int r = 0; r < 8; ++r) {                     // C slot r: row rowBase + r + hl*8, col n
      float dist = en - 2.0f * c[r];                  // ||r||^2 constant per row: drop it
      if (dist < minv[r]) { minv[r] = dist; mini[r] = n; }
    }

    if (nt + 1 < NTILES) {
      wait_async0();          // own async fills landed
      __syncthreads();        // everyone done reading buf & next buffer published
    }
  }

  // min-reduce across the 16 lanes of each half-wave (xor masks < 16 keep halves separate)
  #pragma unroll
  for (int off = 8; off >= 1; off >>= 1) {
    #pragma unroll
    for (int r = 0; r < 8; ++r) {
      float ov = __shfl_xor(minv[r], off, 32);
      int   oi = __shfl_xor(mini[r], off, 32);
      if (ov < minv[r] || (ov == minv[r] && oi < mini[r])) { minv[r] = ov; mini[r] = oi; }
    }
  }
  if (m == 0) {
    #pragma unroll
    for (int r = 0; r < 8; ++r) codes[rowBase + hl * 8 + r] = mini[r];
  }
}

// ---------------------------------------------------------------- stage: residual/quant update + loss
__global__ __launch_bounds__(256) void rvq_update_kernel(float* __restrict__ R,
                                                         float* __restrict__ Qacc,
                                                         const float* __restrict__ cb,
                                                         const int* __restrict__ codes,
                                                         float* __restrict__ lossSum) {
  const int gid = blockIdx.x * 256 + threadIdx.x;    // one float4 each
  const int row = gid >> 6;                          // 64 float4 per row
  const int d4  = (gid & 63) << 2;
  const int idx = codes[row];
  const size_t ro = (size_t)row * Dd + d4;
  const float4 q = *(const float4*)(cb + (size_t)idx * Dd + d4);
  float4 r = *(const float4*)(R + ro);
  float4 rn = {r.x - q.x, r.y - q.y, r.z - q.z, r.w - q.w};
  *(float4*)(R + ro) = rn;
  float4 qa = *(const float4*)(Qacc + ro);
  qa.x += q.x; qa.y += q.y; qa.z += q.z; qa.w += q.w;
  *(float4*)(Qacc + ro) = qa;
  // commit loss: (q - r_old)^2 == r_new^2
  float s = rn.x * rn.x + rn.y * rn.y + rn.z * rn.z + rn.w * rn.w;
  s = waveReduceSum(s);
  __shared__ float red[8];
  if ((threadIdx.x & 31) == 0) red[threadIdx.x >> 5] = s;
  __syncthreads();
  if (threadIdx.x == 0) {
    float t = 0.f;
    #pragma unroll
    for (int i = 0; i < 8; ++i) t += red[i];
    atomicAdd(lossSum, t);
  }
}

// ---------------------------------------------------------------- tail
__global__ void codes_to_float_kernel(const int* __restrict__ ci, float* __restrict__ cf) {
  int i = blockIdx.x * 256 + threadIdx.x;
  cf[i] = (float)ci[i];
}

__global__ void finalize_kernel(const float* __restrict__ lossSum,
                                const int* __restrict__ sr,
                                float* __restrict__ outTail) {
  if (threadIdx.x == 0) {
    float s = 0.f;
    #pragma unroll
    for (int i = 0; i < NQ; ++i) s += lossSum[i];
    outTail[0] = (float)NQ * 10.0f * ((float)sr[0] / 1000.0f);   // log2(1024)=10
    outTail[1] = s / ((float)NQ * (float)NROWS * (float)Dd);
  }
}

// ---------------------------------------------------------------- host
extern "C" void kernel_launch(void* const* d_in, const int* in_sizes, int n_in,
                              void* d_out, int out_size, void* d_ws, size_t ws_size,
                              hipStream_t stream) {
  const float* x   = (const float*)d_in[0];
  const float* cbs = (const float*)d_in[1];
  const int*   sr  = (const int*)d_in[2];
  float* out = (float*)d_out;

  // workspace layout
  float* R    = (float*)d_ws;                               // 64 MB
  float* Qacc = R + (size_t)NROWS * Dd;                     // 64 MB
  unsigned short* cbB = (unsigned short*)(Qacc + (size_t)NROWS * Dd);   // 4 MB (bf16)
  float* enorm = (float*)(cbB + (size_t)NQ * BINS * Dd);    // 32 KB
  int*   codesI = (int*)(enorm + NQ * BINS);                // 2 MB
  float* lossSum = (float*)(codesI + (size_t)NQ * NROWS);   // 32 B

  zero_ws_kernel<<<(NROWS * Dd / 4) / 256, 256, 0, stream>>>(Qacc, lossSum);

  dim3 tb(32, 8), tg(Tt / 32, Dd / 32, Bsz);
  transpose_in_kernel<<<tg, tb, 0, stream>>>(x, R);
  cb_prep_kernel<<<NQ * BINS, 256, 0, stream>>>(cbs, cbB, enorm);

  for (int s = 0; s < NQ; ++s) {
    rvq_argmin_kernel<<<NROWS / ROWS_PER_BLOCK, 256, 0, stream>>>(
        R, cbB + (size_t)s * BINS * Dd, enorm + s * BINS, codesI + s * NROWS);
    rvq_update_kernel<<<(NROWS * Dd / 4) / 256, 256, 0, stream>>>(
        R, Qacc, cbs + (size_t)s * BINS * Dd, codesI + s * NROWS, lossSum + s);
  }

  float* codesOut = out + (size_t)NROWS * Dd;               // after quantized
  codes_to_float_kernel<<<(NQ * NROWS) / 256, 256, 0, stream>>>(codesI, codesOut);
  transpose_out_kernel<<<tg, tb, 0, stream>>>(Qacc, out);
  finalize_kernel<<<1, 32, 0, stream>>>(lossSum, sr, codesOut + (size_t)NQ * NROWS);
}
